// DeformableCrossAttention_75539884802135
// MI455X (gfx1250) — compile-verified, matrix operand order checked by reading, not run on hardware
//
#include <hip/hip_runtime.h>

// ---------------------------------------------------------------------------
// Deformable cross-attention for MI455X (gfx1250, wave32, WMMA, LDS staging)
//   B=8, LQ=2048, C=256, NH=8, NP=4, HD=32, SH=SW=128
// 3 kernels: projection WMMA GEMM -> gather/softmax -> output WMMA GEMM
// Weight panels are staged once per block into LDS, pre-converted to f16 and
// pre-swizzled into WMMA B-fragment order: inner loop reads one fragment as a
// contiguous 32B ds_load. All 8 waves of a block share the panel.
// ---------------------------------------------------------------------------

typedef __attribute__((ext_vector_type(16))) _Float16 v16h;
typedef __attribute__((ext_vector_type(8)))  float    v8f;

#define B_   8
#define LQ_  2048
#define C_   256
#define NH_  8
#define NP_  4
#define HD_  32
#define SH_  128
#define SW_  128
#define M_   (B_ * LQ_)          // 16384 rows
#define NOA_ 96                  // 64 offset cols + 32 attn-logit cols
#define KT_  (C_ / 32)           // 8 K-tiles of 32

// ---- A fragment loader (f32 source -> f16 fragment), wave32 layout ----
// A 16x32 (MxK): lanes 0-15 hold row M=lane with K={0..7,16..23};
//                lanes 16-31 hold row M=lane-16 with K={8..15,24..31}.
__device__ __forceinline__ v16h load_a_f32(const float* __restrict__ A, int lda,
                                           int m0, int k0, int lane) {
  const float* p = A + (size_t)(m0 + (lane & 15)) * lda + k0 + ((lane >> 4) << 3);
  v16h a;
#pragma unroll
  for (int i = 0; i < 8; ++i) {
    a[i]     = (_Float16)p[i];        // K = khalf + i
    a[i + 8] = (_Float16)p[16 + i];   // K = 16 + khalf + i
  }
  return a;
}

// Fragment-order K index for half slot i in lane l (inverse of A/B layout):
__device__ __forceinline__ int frag_kin(int i, int l) {
  return (i & 7) + (((l >> 4) & 1) << 3) + ((i >> 3) << 4);
}

// ---------------------------------------------------------------------------
// Kernel 1: offattn[m, 0:96] = query[m,:] @ [W_off | W_attn] + [b_off | b_attn]
// Block: 8 waves = 8 M-tiles, sharing a 48KB LDS panel of all 96 weight cols
// in WMMA fragment order. Grid: (M/16)/8 = 128 blocks of 256.
// ---------------------------------------------------------------------------
__global__ __launch_bounds__(256) void k_proj_offattn(
    const float* __restrict__ query,
    const float* __restrict__ W_off,  const float* __restrict__ b_off,
    const float* __restrict__ W_attn, const float* __restrict__ b_attn,
    float* __restrict__ offattn) {
  // panel[(ktile*6 + t)*32 + lane][16 halves], 8*6*512 halves = 48 KB
  __shared__ __align__(32) _Float16 bpanel[KT_ * 6 * 32 * 16];

  const int lane = threadIdx.x & 31;
  const int wave = threadIdx.x >> 5;
  const int tid  = threadIdx.x;

  // ---- stage weights into LDS in fragment order (96 halves per thread) ----
#pragma unroll
  for (int s = 0; s < 96; ++s) {
    const int flat  = tid * 96 + s;
    const int i     = flat & 15;
    const int l     = (flat >> 4) & 31;
    const int rest  = flat >> 9;            // fragment id = ktile*6 + t
    const int t     = rest % 6;
    const int ktile = rest / 6;
    const int k     = ktile * 32 + frag_kin(i, l);
    const int nl    = l & 15;
    const float v = (t < 4) ? W_off[(size_t)k * (NH_ * NP_ * 2) + t * 16 + nl]
                            : W_attn[(size_t)k * (NH_ * NP_) + (t - 4) * 16 + nl];
    bpanel[flat] = (_Float16)v;
  }
  __syncthreads();

  const int tm = blockIdx.x * 8 + wave;     // 1024 M-tiles
  const int m0 = tm * 16;
  const int n  = lane & 15;

  v8f acc[6];
#pragma unroll
  for (int t = 0; t < 4; ++t) {             // W_off columns, bias broadcast
    const float bv = b_off[t * 16 + n];
#pragma unroll
    for (int r = 0; r < 8; ++r) acc[t][r] = bv;
  }
#pragma unroll
  for (int t = 0; t < 2; ++t) {             // W_attn columns
    const float bv = b_attn[t * 16 + n];
#pragma unroll
    for (int r = 0; r < 8; ++r) acc[4 + t][r] = bv;
  }

#pragma unroll
  for (int ktile = 0; ktile < KT_; ++ktile) {
    const int k0 = ktile * 32;
    if (k0 + 32 < C_)
      __builtin_prefetch(query + (size_t)(m0 + (lane & 15)) * C_ + k0 + 32, 0, 1);
    const v16h a = load_a_f32(query, C_, m0, k0, lane);
#pragma unroll
    for (int t = 0; t < 6; ++t) {
      const v16h b = *(const v16h*)&bpanel[(size_t)((ktile * 6 + t) * 32 + lane) * 16];
      acc[t] = __builtin_amdgcn_wmma_f32_16x16x32_f16(false, a, false, b,
                                                      (short)0, acc[t], false, false);
    }
  }

  // D layout: lanes 0-15: VGPR r -> (M=r, N=lane); lanes 16-31: (M=8+r, N=lane-16)
  const int mb = (lane >> 4) << 3;
#pragma unroll
  for (int t = 0; t < 6; ++t)
#pragma unroll
    for (int r = 0; r < 8; ++r)
      offattn[(size_t)(m0 + mb + r) * NOA_ + (t * 16 + n)] = acc[t][r];
}

// ---------------------------------------------------------------------------
// Kernel 2: softmax over NP + bilinear gather + attention weighting.
// One wave per (b, q, h); lane = head channel d (HD==32==warpSize).
// Every corner fetch is a coalesced 128B line (channels contiguous in memory);
// the full 128MB memory tensor fits in the 192MB L2 so re-touches are cheap.
// Grid: B*LQ*NH / 8 = 16384 blocks of 256.
// ---------------------------------------------------------------------------
__global__ __launch_bounds__(256) void k_sample(
    const float* __restrict__ memory,     // [B, SH*SW, C]
    const float* __restrict__ boxes,      // [B, LQ, 4]
    const float* __restrict__ offattn,    // [M, 96]
    float* __restrict__ weighted) {       // [M, C]
  const int lane = threadIdx.x & 31;
  const int wave = threadIdx.x >> 5;
  const size_t idx = (size_t)blockIdx.x * 8 + wave;   // flat (b, q, h)
  const int    h   = (int)(idx % NH_);
  const size_t bq  = idx / NH_;                        // b*LQ + q
  const int    b   = (int)(bq / LQ_);

  const float* box = boxes + bq * 4;
  const float cx = box[0], cy = box[1];
  const float bw2 = box[2] * 0.5f, bh2 = box[3] * 0.5f;

  const float* oa = offattn + bq * NOA_;

  // softmax over the NP=4 logits for this head (wave-uniform scalar math)
  float lg[NP_];
#pragma unroll
  for (int p = 0; p < NP_; ++p) lg[p] = oa[64 + h * NP_ + p];
  float mx = fmaxf(fmaxf(lg[0], lg[1]), fmaxf(lg[2], lg[3]));
  float ex[NP_]; float sum = 0.f;
#pragma unroll
  for (int p = 0; p < NP_; ++p) { ex[p] = __expf(lg[p] - mx); sum += ex[p]; }
  const float inv = 1.f / sum;

  // channel-contiguous base: memory[b][y*SW+x][h*HD + lane]
  const float* membase = memory + (size_t)b * (SH_ * SW_) * C_ + h * HD_ + lane;

  auto corner = [&](int xi, int yi, float w) -> float {
    if (xi < 0 || xi >= SW_ || yi < 0 || yi >= SH_) return 0.f;
    return w * membase[(size_t)(yi * SW_ + xi) * C_];
  };

  float acc = 0.f;
#pragma unroll
  for (int p = 0; p < NP_; ++p) {
    const float ap = ex[p] * inv;
    const float ox = oa[(h * NP_ + p) * 2 + 0];
    const float oy = oa[(h * NP_ + p) * 2 + 1];
    // loc = ref_xy + off*ref_wh*0.5 ; pixel coord = loc*W - 0.5
    const float x = (cx + ox * bw2) * (float)SW_ - 0.5f;
    const float y = (cy + oy * bh2) * (float)SH_ - 0.5f;
    const float x0f = floorf(x), y0f = floorf(y);
    const int   x0 = (int)x0f,   y0 = (int)y0f;
    const float wx1 = x - x0f, wy1 = y - y0f;
    const float wx0 = 1.f - wx1, wy0 = 1.f - wy1;
    acc += corner(x0,     y0,     wx0 * wy0 * ap);
    acc += corner(x0 + 1, y0,     wx1 * wy0 * ap);
    acc += corner(x0,     y0 + 1, wx0 * wy1 * ap);
    acc += corner(x0 + 1, y0 + 1, wx1 * wy1 * ap);
  }

  weighted[bq * C_ + h * HD_ + lane] = acc;
}

// ---------------------------------------------------------------------------
// Kernel 3: out = weighted @ W_out + b_out   ([16384x256] x [256x256])
// Block: 8 waves = 8 M-tiles sharing one 64-col N-group; 32KB LDS panel of
// W_out in fragment order. Grid: (M/16)/8 * 4 = 512 blocks of 256.
// ---------------------------------------------------------------------------
__global__ __launch_bounds__(256) void k_proj_out(
    const float* __restrict__ weighted,
    const float* __restrict__ W_out, const float* __restrict__ b_out,
    float* __restrict__ out) {
  // panel[(ktile*4 + t)*32 + lane][16 halves], 8*4*512 halves = 32 KB
  __shared__ __align__(32) _Float16 bpanel[KT_ * 4 * 32 * 16];

  const int lane = threadIdx.x & 31;
  const int wave = threadIdx.x >> 5;
  const int tid  = threadIdx.x;
  const int ng   = blockIdx.x & 3;           // N group: columns [ng*64, +64)
  const int nb   = ng * 64;

  // ---- stage W_out N-group into LDS in fragment order (64 halves/thread) ----
#pragma unroll
  for (int s = 0; s < 64; ++s) {
    const int flat  = tid * 64 + s;
    const int i     = flat & 15;
    const int l     = (flat >> 4) & 31;
    const int rest  = flat >> 9;             // fragment id = ktile*4 + t
    const int t     = rest & 3;
    const int ktile = rest >> 2;
    const int k     = ktile * 32 + frag_kin(i, l);
    bpanel[flat] = (_Float16)W_out[(size_t)k * C_ + nb + t * 16 + (l & 15)];
  }
  __syncthreads();

  const int tm = (blockIdx.x >> 2) * 8 + wave;  // 1024 M-tiles
  const int m0 = tm * 16;
  const int n  = lane & 15;

  v8f acc[4];
#pragma unroll
  for (int t = 0; t < 4; ++t) {
    const float bv = b_out[nb + t * 16 + n];
#pragma unroll
    for (int r = 0; r < 8; ++r) acc[t][r] = bv;
  }

#pragma unroll
  for (int ktile = 0; ktile < KT_; ++ktile) {
    const int k0 = ktile * 32;
    if (k0 + 32 < C_)
      __builtin_prefetch(weighted + (size_t)(m0 + (lane & 15)) * C_ + k0 + 32, 0, 1);
    const v16h a = load_a_f32(weighted, C_, m0, k0, lane);
#pragma unroll
    for (int t = 0; t < 4; ++t) {
      const v16h b = *(const v16h*)&bpanel[(size_t)((ktile * 4 + t) * 32 + lane) * 16];
      acc[t] = __builtin_amdgcn_wmma_f32_16x16x32_f16(false, a, false, b,
                                                      (short)0, acc[t], false, false);
    }
  }

  const int mb = (lane >> 4) << 3;
#pragma unroll
  for (int t = 0; t < 4; ++t)
#pragma unroll
    for (int r = 0; r < 8; ++r)
      out[(size_t)(m0 + mb + r) * C_ + (nb + t * 16 + n)] = acc[t][r];
}

// ---------------------------------------------------------------------------
extern "C" void kernel_launch(void* const* d_in, const int* in_sizes, int n_in,
                              void* d_out, int out_size, void* d_ws, size_t ws_size,
                              hipStream_t stream) {
  const float* query  = (const float*)d_in[0];
  const float* memory = (const float*)d_in[1];
  const float* boxes  = (const float*)d_in[2];
  const float* W_off  = (const float*)d_in[3];
  const float* b_off  = (const float*)d_in[4];
  const float* W_attn = (const float*)d_in[5];
  const float* b_attn = (const float*)d_in[6];
  const float* W_out  = (const float*)d_in[7];
  const float* b_out  = (const float*)d_in[8];
  float* out = (float*)d_out;

  // workspace: offattn [M x 96] f32, then weighted [M x C] f32  (~23 MB)
  float* offattn  = (float*)d_ws;
  float* weighted = (float*)((char*)d_ws + (size_t)M_ * NOA_ * sizeof(float));

  // 1) projection GEMM: 1024 M-tiles, 8 waves/block -> 128 blocks
  k_proj_offattn<<<(M_ / 16) / 8, 256, 0, stream>>>(
      query, W_off, b_off, W_attn, b_attn, offattn);

  // 2) gather: B*LQ*NH waves / 8 -> 16384 blocks
  k_sample<<<(B_ * LQ_ * NH_) / 8, 256, 0, stream>>>(
      memory, boxes, offattn, weighted);

  // 3) output GEMM: 128 M-blocks * 4 N-groups -> 512 blocks
  k_proj_out<<<((M_ / 16) / 8) * 4, 256, 0, stream>>>(
      weighted, W_out, b_out, out);
}
